// CDMoE_22917945491963
// MI455X (gfx1250) — compile-verified, matrix operand order checked by reading, not run on hardware
//
#include <hip/hip_runtime.h>
#include <hip/hip_bf16.h>
#include <stdint.h>

// ---------------------------------------------------------------------------
// CDMoE on gfx1250: bf16 WMMA GEMMs (async global->LDS double-buffered
// pipeline) for MLP + query projection, wave32 VALU/LDS per-token
// product-key retrieval + expert combine.
// ---------------------------------------------------------------------------

typedef __attribute__((ext_vector_type(16))) __bf16 v16bf;
typedef __attribute__((ext_vector_type(8)))  __bf16 v8bf;
typedef __attribute__((ext_vector_type(8)))  float  v8f;
typedef __attribute__((ext_vector_type(4)))  int    v4i;

// AS-qualified vector types for the async-to-LDS builtin:
// signature is (v4i AS1* src, v4i AS3* dst, imm offset, imm cpol).
typedef __attribute__((address_space(1))) v4i gv4i;   // global
typedef __attribute__((address_space(3))) v4i lv4i;   // LDS

#if defined(__has_builtin)
#  if __has_builtin(__builtin_amdgcn_global_load_async_to_lds_b128)
#    define HAVE_ASYNC_LDS 1
#  endif
#  if __has_builtin(__builtin_amdgcn_s_wait_asynccnt)
#    define HAVE_WAIT_ASYNC 1
#  endif
#endif

// 16B global -> LDS copy. Async path uses the gfx1250 ASYNCcnt-tracked DMA;
// fallback is a synchronous vector copy (correct via the same barriers).
__device__ __forceinline__ void cp16_g2l(const __bf16* g, __bf16* l) {
#ifdef HAVE_ASYNC_LDS
    // generic->AS1 via inttoptr (AS0/AS1 share representation); generic shared
    // pointer's low 32 bits are the LDS offset.
    __builtin_amdgcn_global_load_async_to_lds_b128(
        (gv4i*)(uintptr_t)g, (lv4i*)(uint32_t)(uintptr_t)l, 0, 0);
#else
    *(v8bf*)l = *(const v8bf*)g;
#endif
}

template<int N>
__device__ __forceinline__ void wait_async() {
#ifdef HAVE_ASYNC_LDS
#  ifdef HAVE_WAIT_ASYNC
    __builtin_amdgcn_s_wait_asynccnt(N);
#  else
    asm volatile("s_wait_asynccnt %0" :: "i"(N) : "memory");
#  endif
#endif
}

__device__ __forceinline__ v8f wmma_bf16(v16bf a, v16bf b, v8f c) {
    return __builtin_amdgcn_wmma_f32_16x16x32_bf16(
        /*neg_a=*/false, a, /*neg_b=*/false, b,
        /*c_mod=*/(short)0, c, /*reuse_a=*/false, /*reuse_b=*/false);
}

// LDS tiles padded to 40 halves (80B) per row: ds_load_b128 fragment reads
// from 16 rows hit 16 distinct bank groups (stride 20 dwords mod 64 banks).
#define LDS_LD 40

// Stage a 128x32 bf16 tile (512 x 16B chunks, 2 per thread).
__device__ __forceinline__ void stageA128(const __bf16* __restrict__ A, int K,
                                          int mBase, int k0,
                                          __bf16* dst, int tid) {
    #pragma unroll
    for (int t = 0; t < 2; t++) {
        int id = tid + t * 256;
        int row = id >> 2, c = id & 3;
        cp16_g2l(A + (size_t)(mBase + row) * K + k0 + c * 8,
                 dst + row * LDS_LD + c * 8);
    }
}
// Stage a 64x32 bf16 tile (256 x 16B chunks, 1 per thread).
__device__ __forceinline__ void stageB64(const __bf16* __restrict__ B, int K,
                                         int nBase, int k0,
                                         __bf16* dst, int tid) {
    int row = tid >> 2, c = tid & 3;
    cp16_g2l(B + (size_t)(nBase + row) * K + k0 + c * 8,
             dst + row * LDS_LD + c * 8);
}

// A/B fragment from an LDS tile, ISA 7.12.2 layout:
// halves 0..7 -> K = hi*8+{0..7}, halves 8..15 -> K = 16+hi*8+{0..7}.
__device__ __forceinline__ v16bf frag_lds(const __bf16* t, int row, int hi) {
    union { v16bf v; v8bf h[2]; } u;
    const __bf16* p = t + row * LDS_LD + hi * 8;
    u.h[0] = *(const v8bf*)(p);
    u.h[1] = *(const v8bf*)(p + 16);
    return u.v;
}

// ---------------------------------------------------------------------------
// Elementwise fp32 -> bf16
// ---------------------------------------------------------------------------
__global__ void k_f32_to_bf16(const float* __restrict__ in,
                              __bf16* __restrict__ out, int n) {
    int i = blockIdx.x * blockDim.x + threadIdx.x;
    if (i < n) out[i] = (__bf16)in[i];
}

// ---------------------------------------------------------------------------
// Transposed fp32 -> bf16 conversion (W[R,C] -> Wt[C,R]) via LDS tile.
// ---------------------------------------------------------------------------
__global__ __launch_bounds__(256)
void k_transpose_to_bf16(const float* __restrict__ W, __bf16* __restrict__ Wt,
                         int R, int C) {
    __shared__ __bf16 tile[32][33];
    int c0 = blockIdx.x * 32, r0 = blockIdx.y * 32;
    int tx = threadIdx.x, ty = threadIdx.y;
    #pragma unroll
    for (int j = 0; j < 32; j += 8)
        tile[ty + j][tx] = (__bf16)W[(size_t)(r0 + ty + j) * C + (c0 + tx)];
    __syncthreads();
    #pragma unroll
    for (int j = 0; j < 32; j += 8)
        Wt[(size_t)(c0 + ty + j) * R + (r0 + tx)] = tile[tx][ty + j];
}

// ---------------------------------------------------------------------------
// Generic WMMA GEMM: C[M,N] = A[M,K] x Bt[N,K]^T (+bias).
// Block = 8 waves, block tile 128x64, wave tile 32x32.
// Double-buffered async global->LDS staging; 3 chunks in flight per thread.
// ---------------------------------------------------------------------------
template<bool HAS_BIAS, bool OUT_BF16>
__global__ __launch_bounds__(256)
void k_wmma_gemm(const __bf16* __restrict__ A, const __bf16* __restrict__ Bt,
                 const float* __restrict__ bias, void* __restrict__ Cout,
                 int M, int N, int K) {
    __shared__ __align__(16) __bf16 tA[2][128 * LDS_LD];
    __shared__ __align__(16) __bf16 tB[2][64 * LDS_LD];
    const int tid = threadIdx.x;
    const int lane = tid & 31, wid = tid >> 5;
    const int lo = lane & 15, hi = lane >> 4;
    const int mBlk = blockIdx.x * 128, nBlk = blockIdx.y * 64;
    const int wm = (wid & 3) * 32, wn = (wid >> 2) * 32;

    v8f acc[2][2] = {};
    stageA128(A, K, mBlk, 0, tA[0], tid);
    stageB64(Bt, K, nBlk, 0, tB[0], tid);

    int buf = 0;
    for (int k0 = 0; k0 < K; k0 += 32) {
        if (k0 + 32 < K) {
            stageA128(A, K, mBlk, k0 + 32, tA[buf ^ 1], tid);
            stageB64(Bt, K, nBlk, k0 + 32, tB[buf ^ 1], tid);
            wait_async<3>();   // current buffer complete, next 3 in flight
        } else {
            wait_async<0>();
        }
        __syncthreads();
        v16bf a0 = frag_lds(tA[buf], wm + lo,      hi);
        v16bf a1 = frag_lds(tA[buf], wm + 16 + lo, hi);
        v16bf b0 = frag_lds(tB[buf], wn + lo,      hi);
        v16bf b1 = frag_lds(tB[buf], wn + 16 + lo, hi);
        acc[0][0] = wmma_bf16(a0, b0, acc[0][0]);
        acc[0][1] = wmma_bf16(a0, b1, acc[0][1]);
        acc[1][0] = wmma_bf16(a1, b0, acc[1][0]);
        acc[1][1] = wmma_bf16(a1, b1, acc[1][1]);
        __syncthreads();       // all waves done reading before refill
        buf ^= 1;
    }

    #pragma unroll
    for (int mi = 0; mi < 2; mi++)
        #pragma unroll
        for (int ni = 0; ni < 2; ni++) {
            int n = nBlk + wn + ni * 16 + lo;
            float bv = HAS_BIAS ? bias[n] : 0.f;
            #pragma unroll
            for (int r = 0; r < 8; r++) {
                int m = mBlk + wm + mi * 16 + hi * 8 + r;
                float v = acc[mi][ni][r] + bv;
                if (OUT_BF16) ((__bf16*)Cout)[(size_t)m * N + n] = (__bf16)v;
                else          ((float*)Cout)[(size_t)m * N + n] = v;
            }
        }
}

// ---------------------------------------------------------------------------
// Fused gate/up GEMM: act = silu(X*Wg + bg) * (X*Wu + bu), bf16 out.
// A fragments reused across both weight streams (8 WMMA / k-step).
// 4 staged chunks per thread per k-step.
// ---------------------------------------------------------------------------
__global__ __launch_bounds__(256)
void k_gemm_gateup(const __bf16* __restrict__ X,
                   const __bf16* __restrict__ WgT, const __bf16* __restrict__ WuT,
                   const float* __restrict__ bg, const float* __restrict__ bu,
                   __bf16* __restrict__ act, int M, int N, int K) {
    __shared__ __align__(16) __bf16 tA[2][128 * LDS_LD];
    __shared__ __align__(16) __bf16 tG[2][64 * LDS_LD];
    __shared__ __align__(16) __bf16 tU[2][64 * LDS_LD];
    const int tid = threadIdx.x;
    const int lane = tid & 31, wid = tid >> 5;
    const int lo = lane & 15, hi = lane >> 4;
    const int mBlk = blockIdx.x * 128, nBlk = blockIdx.y * 64;
    const int wm = (wid & 3) * 32, wn = (wid >> 2) * 32;

    v8f ag[2][2] = {};
    v8f au[2][2] = {};
    stageA128(X, K, mBlk, 0, tA[0], tid);
    stageB64(WgT, K, nBlk, 0, tG[0], tid);
    stageB64(WuT, K, nBlk, 0, tU[0], tid);

    int buf = 0;
    for (int k0 = 0; k0 < K; k0 += 32) {
        if (k0 + 32 < K) {
            stageA128(X, K, mBlk, k0 + 32, tA[buf ^ 1], tid);
            stageB64(WgT, K, nBlk, k0 + 32, tG[buf ^ 1], tid);
            stageB64(WuT, K, nBlk, k0 + 32, tU[buf ^ 1], tid);
            wait_async<4>();
        } else {
            wait_async<0>();
        }
        __syncthreads();
        v16bf a0 = frag_lds(tA[buf], wm + lo,      hi);
        v16bf a1 = frag_lds(tA[buf], wm + 16 + lo, hi);
        v16bf g0 = frag_lds(tG[buf], wn + lo,      hi);
        v16bf g1 = frag_lds(tG[buf], wn + 16 + lo, hi);
        v16bf u0 = frag_lds(tU[buf], wn + lo,      hi);
        v16bf u1 = frag_lds(tU[buf], wn + 16 + lo, hi);
        ag[0][0] = wmma_bf16(a0, g0, ag[0][0]);
        ag[0][1] = wmma_bf16(a0, g1, ag[0][1]);
        ag[1][0] = wmma_bf16(a1, g0, ag[1][0]);
        ag[1][1] = wmma_bf16(a1, g1, ag[1][1]);
        au[0][0] = wmma_bf16(a0, u0, au[0][0]);
        au[0][1] = wmma_bf16(a0, u1, au[0][1]);
        au[1][0] = wmma_bf16(a1, u0, au[1][0]);
        au[1][1] = wmma_bf16(a1, u1, au[1][1]);
        __syncthreads();
        buf ^= 1;
    }

    #pragma unroll
    for (int mi = 0; mi < 2; mi++)
        #pragma unroll
        for (int ni = 0; ni < 2; ni++) {
            int n = nBlk + wn + ni * 16 + lo;
            float bgv = bg[n], buv = bu[n];
            #pragma unroll
            for (int r = 0; r < 8; r++) {
                int m = mBlk + wm + mi * 16 + hi * 8 + r;
                float g = ag[mi][ni][r] + bgv;
                float u = au[mi][ni][r] + buv;
                float s = g / (1.f + expf(-g));    // silu
                act[(size_t)m * N + n] = (__bf16)(s * u);
            }
        }
}

// ---------------------------------------------------------------------------
// Per-token product-key retrieval + expert combine (adds into out).
// One block (256 thr = 8 wave32) per token.
// ---------------------------------------------------------------------------
__global__ __launch_bounds__(256)
void k_experts(const float* __restrict__ x_all,   // [4096,1024] fp32 hidden
               const float* __restrict__ q_all,   // [4096,512]  fp32 queries
               const float* __restrict__ keys,    // [4,64,2,64]
               const float* __restrict__ down,    // [4096,1024]
               const float* __restrict__ up,      // [4096,1024]
               float* __restrict__ out) {         // [4096,1024] (has MLP term)
    const int token = blockIdx.x;
    const int tid = threadIdx.x;

    __shared__ float sx[1024];
    __shared__ float sq[512];
    __shared__ float sim[512];        // [p][h][k] = p*256 + h*64 + k
    __shared__ float hs[2][4][8];
    __shared__ int   hix[2][4][8];
    __shared__ float sprob[32];       // [h*8 + j]
    __shared__ int   sidx[32];
    __shared__ float sw[32];

    for (int i = tid; i < 1024; i += 256) sx[i] = x_all[(size_t)token * 1024 + i];
    for (int i = tid; i < 512;  i += 256) sq[i] = q_all[(size_t)token * 512 + i];
    __syncthreads();

    // sim[p,h,k] = q[p,h,:] . keys[h,k,p,:]   (512 dots of 64)
    for (int i = tid; i < 512; i += 256) {
        int p = i >> 8, h = (i >> 6) & 3, k = i & 63;
        const float* kp = keys + (((size_t)h * 64 + k) * 2 + p) * 64;
        const float* qp = sq + p * 256 + h * 64;
        float s = 0.f;
        #pragma unroll 8
        for (int n = 0; n < 64; n++) s += qp[n] * kp[n];
        sim[i] = s;
    }
    __syncthreads();

    // stage 1: per (p,h) top-8 of 64
    if (tid < 8) {
        int p = tid >> 2, h = tid & 3;
        const float* sp = sim + p * 256 + h * 64;
        unsigned long long used = 0ULL;
        for (int j = 0; j < 8; j++) {
            float best = -3.4e38f; int bi = 0;
            for (int k = 0; k < 64; k++) {
                if ((used >> k) & 1ULL) continue;
                float v = sp[k];
                if (v > best) { best = v; bi = k; }
            }
            used |= 1ULL << bi;
            hs[p][h][j] = best; hix[p][h][j] = bi;
        }
    }
    __syncthreads();

    // stage 2: per h, top-8 of the 8x8 combined grid + softmax
    if (tid < 4) {
        int h = tid;
        float cs[8]; int ca[8], cb[8];
        unsigned long long used = 0ULL;
        for (int j = 0; j < 8; j++) {
            float best = -3.4e38f; int bi = 0;
            for (int c = 0; c < 64; c++) {
                if ((used >> c) & 1ULL) continue;
                float v = hs[0][h][c >> 3] + hs[1][h][c & 7];
                if (v > best) { best = v; bi = c; }
            }
            used |= 1ULL << bi;
            cs[j] = best; ca[j] = bi >> 3; cb[j] = bi & 7;
        }
        float mx = cs[0];                 // sorted desc -> cs[0] is max
        float den = 0.f, e[8];
        for (int j = 0; j < 8; j++) { e[j] = expf(cs[j] - mx); den += e[j]; }
        float inv = 1.f / den;
        for (int j = 0; j < 8; j++) {
            sprob[h * 8 + j] = e[j] * inv;
            sidx[h * 8 + j]  = hix[0][h][ca[j]] * 64 + hix[1][h][cb[j]];
        }
    }
    __syncthreads();

    // w_e = silu(x . down_embed[idx_e]) * prob_e  -- wave32 per expert
    const int wv = tid >> 5, ln = tid & 31;
    for (int e = wv; e < 32; e += 8) {
        const float* de = down + (size_t)sidx[e] * 1024;
        float s = 0.f;
        for (int i = ln; i < 1024; i += 32) s += sx[i] * de[i];
        #pragma unroll
        for (int off = 16; off > 0; off >>= 1) s += __shfl_xor(s, off, 32);
        if (ln == 0) {
            float sg = 1.f / (1.f + expf(-s));
            sw[e] = s * sg * sprob[e];
        }
    }
    __syncthreads();

    // out[token,:] += sum_e w_e * up_embed[idx_e,:]   (float4 per thread)
    float4 acc = make_float4(0.f, 0.f, 0.f, 0.f);
    for (int e = 0; e < 32; e++) {
        float we = sw[e];
        const float4* ur = (const float4*)(up + (size_t)sidx[e] * 1024);
        float4 u4 = ur[tid];
        acc.x += we * u4.x; acc.y += we * u4.y;
        acc.z += we * u4.z; acc.w += we * u4.w;
    }
    float4* op = (float4*)(out + (size_t)token * 1024);
    float4 m = op[tid];
    m.x += acc.x; m.y += acc.y; m.z += acc.z; m.w += acc.w;
    op[tid] = m;
}

// ---------------------------------------------------------------------------
extern "C" void kernel_launch(void* const* d_in, const int* in_sizes, int n_in,
                              void* d_out, int out_size, void* d_ws, size_t ws_size,
                              hipStream_t stream) {
    (void)in_sizes; (void)n_in; (void)out_size; (void)ws_size;
    const float* hidden = (const float*)d_in[0];
    const float* Wg  = (const float*)d_in[1];
    const float* bg  = (const float*)d_in[2];
    const float* Wu  = (const float*)d_in[3];
    const float* bu  = (const float*)d_in[4];
    const float* Wd  = (const float*)d_in[5];
    const float* bd  = (const float*)d_in[6];
    const float* Wq  = (const float*)d_in[7];
    const float* keys = (const float*)d_in[8];
    const float* down = (const float*)d_in[9];
    const float* up   = (const float*)d_in[10];
    float* out = (float*)d_out;

    constexpr int Mtok = 4096;   // B*T
    constexpr int D    = 1024;
    constexpr int DCD  = 4096;
    constexpr int QN   = 512;    // RET*H

    char* ws = (char*)d_ws;
    size_t off = 0;
    auto take = [&](size_t bytes) -> void* {
        void* p = ws + off;
        off = (off + bytes + 255) & ~(size_t)255;
        return p;
    };
    __bf16* Xbf = (__bf16*)take((size_t)Mtok * D * 2);
    __bf16* WgT = (__bf16*)take((size_t)D * DCD * 2);
    __bf16* WuT = (__bf16*)take((size_t)D * DCD * 2);
    __bf16* WdT = (__bf16*)take((size_t)DCD * D * 2);
    __bf16* WqT = (__bf16*)take((size_t)D * QN * 2);
    __bf16* act = (__bf16*)take((size_t)Mtok * DCD * 2);
    float*  qws = (float*)take((size_t)Mtok * QN * 4);

    // 1) precision conversion + weight transposes
    k_f32_to_bf16<<<(Mtok * D + 255) / 256, 256, 0, stream>>>(hidden, Xbf, Mtok * D);
    dim3 tb(32, 8);
    k_transpose_to_bf16<<<dim3(DCD / 32, D / 32), tb, 0, stream>>>(Wg, WgT, D, DCD);
    k_transpose_to_bf16<<<dim3(DCD / 32, D / 32), tb, 0, stream>>>(Wu, WuT, D, DCD);
    k_transpose_to_bf16<<<dim3(D / 32, DCD / 32), tb, 0, stream>>>(Wd, WdT, DCD, D);
    k_transpose_to_bf16<<<dim3(QN / 32, D / 32), tb, 0, stream>>>(Wq, WqT, D, QN);

    // 2) query projection (WMMA, fp32 out)
    k_wmma_gemm<false, false><<<dim3(Mtok / 128, QN / 64), 256, 0, stream>>>(
        Xbf, WqT, nullptr, qws, Mtok, QN, D);

    // 3) fused gate/up GEMM -> silu(g)*u (WMMA, bf16 out)
    k_gemm_gateup<<<dim3(Mtok / 128, DCD / 64), 256, 0, stream>>>(
        Xbf, WgT, WuT, bg, bu, act, Mtok, DCD, D);

    // 4) down projection + bias -> MLP term in d_out (WMMA, fp32 out)
    k_wmma_gemm<true, false><<<dim3(Mtok / 128, D / 64), 256, 0, stream>>>(
        act, WdT, bd, out, Mtok, D, DCD);

    // 5) product-key retrieval + expert combine, adds into d_out
    k_experts<<<Mtok, 256, 0, stream>>>(hidden, qws, keys, down, up, out);
}
